// WindowCrossAttention_76115410420090
// MI455X (gfx1250) — matrix operand order repaired and checked by reading, not compiled, not run on hardware
//
#include <hip/hip_runtime.h>
#include <hip/hip_bf16.h>

#define B_TOT 4096
#define NTOK  64
#define DIMC  512
#define NH    8
#define HDIM  64
#define NWIN  256

// padded LDS strides (elements)
#define XS_ST 520   // 1040B rows: breaks the 16-way bank conflict of stride-1024B
#define P_ST  72
#define V_ST  72

typedef __attribute__((ext_vector_type(16))) __bf16 v16bf;
typedef __attribute__((ext_vector_type(8)))  __bf16 v8bf;
typedef __attribute__((ext_vector_type(4)))  __bf16 v4bf;
typedef __attribute__((ext_vector_type(8)))  float  v8f;
typedef __attribute__((ext_vector_type(4)))  float  v4f;

__device__ __forceinline__ __bf16 f2bf(float f) {
  union { float f; unsigned u; } c{f};
  unsigned r = (c.u + 0x7FFFu + ((c.u >> 16) & 1u)) >> 16;   // RNE
  union { unsigned short s; __bf16 b; } o;
  o.s = (unsigned short)r;
  return o.b;
}

__device__ __forceinline__ v8bf cvt8(const float* p) {
  v4f a = *(const v4f*)p;
  v4f b = *(const v4f*)(p + 4);
  v8bf r;
  r[0]=f2bf(a[0]); r[1]=f2bf(a[1]); r[2]=f2bf(a[2]); r[3]=f2bf(a[3]);
  r[4]=f2bf(b[0]); r[5]=f2bf(b[1]); r[6]=f2bf(b[2]); r[7]=f2bf(b[3]);
  return r;
}

__device__ __forceinline__ v16bf cat16(v8bf lo, v8bf hi) {
  return __builtin_shufflevector(lo, hi, 0,1,2,3,4,5,6,7,8,9,10,11,12,13,14,15);
}

__device__ __forceinline__ v8f zero8() {
  v8f z;
#pragma unroll
  for (int i = 0; i < 8; ++i) z[i] = 0.0f;
  return z;
}

#define WMMA_BF16(A, B, C) \
  __builtin_amdgcn_wmma_f32_16x16x32_bf16(false, (A), false, (B), (short)0, (C), false, false)

// ---------------------------------------------------------------------------
// Prep: fp32 weights -> bf16, gather relative-position bias -> [H,64,64] fp32
// ---------------------------------------------------------------------------
__global__ void swin_prep_kernel(const float* __restrict__ qkv_w,
                                 const float* __restrict__ proj_w,
                                 const float* __restrict__ table,
                                 const int*   __restrict__ idx,
                                 __bf16* __restrict__ wqkv,
                                 __bf16* __restrict__ wproj,
                                 float*  __restrict__ bias) {
  int t = blockIdx.x * blockDim.x + threadIdx.x;   // grid covers 512*512 = 262144
  wqkv[t]  = f2bf(qkv_w[t]);
  wproj[t] = f2bf(proj_w[t]);
  if (t < NH * NTOK * NTOK) {
    int h  = t >> 12;          // head
    int nm = t & 4095;         // n*64+m
    bias[t] = table[idx[nm] * NH + h];
  }
}

// ---------------------------------------------------------------------------
// Fused: q-proj -> QK^T+bias+mask -> softmax -> PV -> out-proj, one window/block
// ---------------------------------------------------------------------------
__global__ __launch_bounds__(256, 1)
void swin_attn_kernel(const float* __restrict__ x,
                      const float* __restrict__ pk,
                      const float* __restrict__ pv,
                      const float* __restrict__ mask,
                      const float* __restrict__ qkv_b,
                      const float* __restrict__ proj_b,
                      const __bf16* __restrict__ wqkv,
                      const __bf16* __restrict__ wproj,
                      const float* __restrict__ bias,
                      float* __restrict__ out) {
  extern __shared__ char smem[];
  __bf16* xs  = (__bf16*)smem;                 // [64][XS_ST] bf16; reused as O stage
  __bf16* qs  = xs + NTOK * XS_ST;             // [64][XS_ST] bf16
  float*  ms  = (float*)(qs + NTOK * XS_ST);   // [64][64] fp32 mask
  __bf16* pvb = (__bf16*)(ms + NTOK * NTOK);   // per-wave: P [64][P_ST] + Vt [64][V_ST]

  const int tid  = threadIdx.x;
  const int wave = tid >> 5;        // 0..7 == head id == output column strip
  const int lane = tid & 31;
  const int l15  = lane & 15;
  const int hi   = lane >> 4;
  const int b    = blockIdx.x;

  // ---- Stage A: x[b] -> LDS bf16 ; mask[b%NW] -> LDS ----
  const float* xb = x + (size_t)b * (NTOK * DIMC);
#pragma unroll
  for (int it = 0; it < 32; ++it) {
    int flat = (it * 256 + tid) * 4;
    int row  = flat >> 9;
    int col  = flat & 511;
    v4f v = *(const v4f*)(xb + flat);
    v4bf o; o[0]=f2bf(v[0]); o[1]=f2bf(v[1]); o[2]=f2bf(v[2]); o[3]=f2bf(v[3]);
    *(v4bf*)(xs + row * XS_ST + col) = o;
  }
  const float* mb = mask + (size_t)(b & (NWIN - 1)) * (NTOK * NTOK);
#pragma unroll
  for (int it = 0; it < 4; ++it) {
    int flat = (it * 256 + tid) * 4;
    *(v4f*)(ms + flat) = *(const v4f*)(mb + flat);
  }
  __syncthreads();

  v8f acc[4][4];

  // ---- Stage B: q strip = scale * (x @ qkv_w^T + qkv_b), cols [64w,64w+64) ----
#pragma unroll
  for (int i = 0; i < 4; ++i)
#pragma unroll
    for (int j = 0; j < 4; ++j) acc[i][j] = zero8();

  {
    const __bf16* wbase = wqkv + (size_t)(wave * 64) * DIMC;
#pragma unroll 4
    for (int kb = 0; kb < DIMC; kb += 32) {
      v16bf afr[4];
#pragma unroll
      for (int i = 0; i < 4; ++i) {
        const __bf16* p = xs + (i * 16 + l15) * XS_ST + kb + hi * 16;
        afr[i] = cat16(*(const v8bf*)p, *(const v8bf*)(p + 8));
      }
#pragma unroll
      for (int j = 0; j < 4; ++j) {
        const __bf16* p = wbase + (size_t)(j * 16 + l15) * DIMC + kb + hi * 16;
        v16bf bfr = cat16(*(const v8bf*)p, *(const v8bf*)(p + 8));
#pragma unroll
        for (int i = 0; i < 4; ++i) acc[i][j] = WMMA_BF16(afr[i], bfr, acc[i][j]);
      }
    }
  }
  {
    const float scale = 0.125f;   // HD^-0.5
#pragma unroll
    for (int j = 0; j < 4; ++j) {
      float qb = qkv_b[wave * 64 + j * 16 + l15];
#pragma unroll
      for (int i = 0; i < 4; ++i)
#pragma unroll
        for (int r = 0; r < 8; ++r) {
          int M = i * 16 + hi * 8 + r;
          qs[M * XS_ST + wave * 64 + j * 16 + l15] = f2bf((acc[i][j][r] + qb) * scale);
        }
    }
  }
  __syncthreads();

  // ---- Stage C: S = q_h @ k_h^T (k converted bf16 in-registers from global) ----
#pragma unroll
  for (int i = 0; i < 4; ++i)
#pragma unroll
    for (int j = 0; j < 4; ++j) acc[i][j] = zero8();

  {
    const float* kbase = pk + ((size_t)b * NH + wave) * (NTOK * HDIM);
#pragma unroll
    for (int kb = 0; kb < HDIM; kb += 32) {
      v16bf afr[4];
#pragma unroll
      for (int i = 0; i < 4; ++i) {
        const __bf16* p = qs + (i * 16 + l15) * XS_ST + wave * 64 + kb + hi * 16;
        afr[i] = cat16(*(const v8bf*)p, *(const v8bf*)(p + 8));
      }
#pragma unroll
      for (int j = 0; j < 4; ++j) {
        const float* p = kbase + (j * 16 + l15) * HDIM + kb + hi * 16;
        v16bf bfr = cat16(cvt8(p), cvt8(p + 8));
#pragma unroll
        for (int i = 0; i < 4; ++i) acc[i][j] = WMMA_BF16(afr[i], bfr, acc[i][j]);
      }
    }
  }

  // ---- Stage D: +bias+mask, softmax (per-row: 4 j-tiles in-lane + shfl_xor) ----
  {
    const float* bh = bias + wave * (NTOK * NTOK);
    __bf16* pb = pvb + wave * (NTOK * (P_ST + V_ST));
#pragma unroll
    for (int i = 0; i < 4; ++i) {
      float rmax[8], rsum[8];
#pragma unroll
      for (int r = 0; r < 8; ++r) rmax[r] = -1e30f;
#pragma unroll
      for (int j = 0; j < 4; ++j) {
        int Nc = j * 16 + l15;
#pragma unroll
        for (int r = 0; r < 8; ++r) {
          int M = i * 16 + hi * 8 + r;
          float v = acc[i][j][r] + bh[M * 64 + Nc] + ms[M * 64 + Nc];
          acc[i][j][r] = v;
          rmax[r] = fmaxf(rmax[r], v);
        }
      }
#pragma unroll
      for (int r = 0; r < 8; ++r) {
#pragma unroll
        for (int m = 1; m <= 8; m <<= 1)
          rmax[r] = fmaxf(rmax[r], __shfl_xor(rmax[r], m, 32));
        rsum[r] = 0.0f;
      }
#pragma unroll
      for (int j = 0; j < 4; ++j)
#pragma unroll
        for (int r = 0; r < 8; ++r) {
          float e = __expf(acc[i][j][r] - rmax[r]);
          acc[i][j][r] = e;
          rsum[r] += e;
        }
#pragma unroll
      for (int r = 0; r < 8; ++r) {
#pragma unroll
        for (int m = 1; m <= 8; m <<= 1)
          rsum[r] += __shfl_xor(rsum[r], m, 32);
        rsum[r] = 1.0f / rsum[r];
      }
#pragma unroll
      for (int j = 0; j < 4; ++j) {
        int Nc = j * 16 + l15;
#pragma unroll
        for (int r = 0; r < 8; ++r) {
          int M = i * 16 + hi * 8 + r;
          pb[M * P_ST + Nc] = f2bf(acc[i][j][r] * rsum[r]);
        }
      }
    }
  }

  // ---- Stage E: v_h -> LDS transposed bf16 (makes PV's B-frags contiguous) ----
  {
    const float* vb = pv + ((size_t)b * NH + wave) * (NTOK * HDIM);
    __bf16* vt = pvb + wave * (NTOK * (P_ST + V_ST)) + NTOK * P_ST;
#pragma unroll
    for (int it = 0; it < 32; ++it) {
      int flat = (it * 32 + lane) * 4;
      int m = flat >> 6;
      int d = flat & 63;
      v4f v = *(const v4f*)(vb + flat);
      vt[(d + 0) * V_ST + m] = f2bf(v[0]);
      vt[(d + 1) * V_ST + m] = f2bf(v[1]);
      vt[(d + 2) * V_ST + m] = f2bf(v[2]);
      vt[(d + 3) * V_ST + m] = f2bf(v[3]);
    }
  }
  __syncthreads();

  // ---- Stage F: O = P @ V ----
#pragma unroll
  for (int i = 0; i < 4; ++i)
#pragma unroll
    for (int j = 0; j < 4; ++j) acc[i][j] = zero8();
  {
    const __bf16* pb = pvb + wave * (NTOK * (P_ST + V_ST));
    const __bf16* vt = pb + NTOK * P_ST;
#pragma unroll
    for (int kb = 0; kb < NTOK; kb += 32) {
      v16bf afr[4];
#pragma unroll
      for (int i = 0; i < 4; ++i) {
        const __bf16* p = pb + (i * 16 + l15) * P_ST + kb + hi * 16;
        afr[i] = cat16(*(const v8bf*)p, *(const v8bf*)(p + 8));
      }
#pragma unroll
      for (int j = 0; j < 4; ++j) {
        const __bf16* p = vt + (j * 16 + l15) * V_ST + kb + hi * 16;
        v16bf bfr = cat16(*(const v8bf*)p, *(const v8bf*)(p + 8));
#pragma unroll
        for (int i = 0; i < 4; ++i) acc[i][j] = WMMA_BF16(afr[i], bfr, acc[i][j]);
      }
    }
  }

  // ---- stage O into LDS (reuse xs region) as [64][512] bf16 ----
  {
    __bf16* os = xs;
#pragma unroll
    for (int j = 0; j < 4; ++j)
#pragma unroll
      for (int i = 0; i < 4; ++i)
#pragma unroll
        for (int r = 0; r < 8; ++r) {
          int M = i * 16 + hi * 8 + r;
          os[M * XS_ST + wave * 64 + j * 16 + l15] = f2bf(acc[i][j][r]);
        }
  }
  __syncthreads();

  // ---- Stage G: out = O @ proj_w^T + proj_b, fp32 stores ----
#pragma unroll
  for (int i = 0; i < 4; ++i)
#pragma unroll
    for (int j = 0; j < 4; ++j) acc[i][j] = zero8();
  {
    const __bf16* os = xs;
    const __bf16* wbase = wproj + (size_t)(wave * 64) * DIMC;
#pragma unroll 4
    for (int kb = 0; kb < DIMC; kb += 32) {
      v16bf afr[4];
#pragma unroll
      for (int i = 0; i < 4; ++i) {
        const __bf16* p = os + (i * 16 + l15) * XS_ST + kb + hi * 16;
        afr[i] = cat16(*(const v8bf*)p, *(const v8bf*)(p + 8));
      }
#pragma unroll
      for (int j = 0; j < 4; ++j) {
        const __bf16* p = wbase + (size_t)(j * 16 + l15) * DIMC + kb + hi * 16;
        v16bf bfr = cat16(*(const v8bf*)p, *(const v8bf*)(p + 8));
#pragma unroll
        for (int i = 0; i < 4; ++i) acc[i][j] = WMMA_BF16(afr[i], bfr, acc[i][j]);
      }
    }
  }
  {
    float* ob = out + (size_t)b * (NTOK * DIMC);
#pragma unroll
    for (int j = 0; j < 4; ++j) {
      float pbv = proj_b[wave * 64 + j * 16 + l15];
#pragma unroll
      for (int i = 0; i < 4; ++i)
#pragma unroll
        for (int r = 0; r < 8; ++r) {
          int M = i * 16 + hi * 8 + r;
          ob[M * DIMC + wave * 64 + j * 16 + l15] = acc[i][j][r] + pbv;
        }
    }
  }
}

// ---------------------------------------------------------------------------
extern "C" void kernel_launch(void* const* d_in, const int* in_sizes, int n_in,
                              void* d_out, int out_size, void* d_ws, size_t ws_size,
                              hipStream_t stream) {
  (void)in_sizes; (void)n_in; (void)out_size; (void)ws_size;
  const float* x      = (const float*)d_in[0];
  const float* pk     = (const float*)d_in[1];
  const float* pv     = (const float*)d_in[2];
  const float* mask   = (const float*)d_in[3];
  const float* table  = (const float*)d_in[4];
  const int*   idx    = (const int*)d_in[5];
  const float* qkv_w  = (const float*)d_in[6];
  const float* qkv_b  = (const float*)d_in[7];
  const float* proj_w = (const float*)d_in[8];
  const float* proj_b = (const float*)d_in[9];
  float* out = (float*)d_out;

  __bf16* wqkv  = (__bf16*)d_ws;
  __bf16* wproj = wqkv + DIMC * DIMC;
  float*  bias  = (float*)(wproj + DIMC * DIMC);

  swin_prep_kernel<<<(DIMC * DIMC) / 256, 256, 0, stream>>>(
      qkv_w, proj_w, table, idx, wqkv, wproj, bias);

  size_t lds = (size_t)2 * NTOK * XS_ST * 2      // xs + qs (bf16)
             + (size_t)NTOK * NTOK * 4           // mask (fp32)
             + (size_t)NH * NTOK * (P_ST + V_ST) * 2;  // per-wave P + Vt
  swin_attn_kernel<<<B_TOT, 256, lds, stream>>>(
      x, pk, pv, mask, qkv_b, proj_b, wqkv, wproj, bias, out);

  const size_t out_elems = (size_t)B_TOT * NTOK * DIMC;       // 134217728
  const size_t kv_elems  = (size_t)B_TOT * NH * NTOK * HDIM;  // 134217728
  hipMemcpyAsync(out + out_elems, pk, kv_elems * sizeof(float),
                 hipMemcpyDeviceToDevice, stream);
  hipMemcpyAsync(out + out_elems + kv_elems, pv, kv_elems * sizeof(float),
                 hipMemcpyDeviceToDevice, stream);
}
